// Seq2Seq_24876450578602
// MI455X (gfx1250) — matrix operand order repaired and checked by reading, not compiled, not run on hardware
//
#include <hip/hip_runtime.h>
#include <stdint.h>

// Problem constants (match reference)
static constexpr int V = 32000;
static constexpr int E = 512;
static constexpr int H = 1024;
static constexpr int S = 512;
static constexpr int T = 512;
static constexpr int G4H = 4 * H;      // 4096 gates
static constexpr int NVB2 = V / 32;    // 1000 vocab blocks of 32

// ---- CDNA5 WMMA vector types ----
typedef __attribute__((ext_vector_type(16))) __bf16    v16bf;
typedef __attribute__((ext_vector_type(8)))  float     v8f;
typedef __attribute__((ext_vector_type(8)))  uint32_t  v8u;

// ---------- helpers ----------
__device__ __forceinline__ uint32_t f32_to_bf16_bits(float f) {
    uint32_t u = __float_as_uint(f);
    uint32_t r = (u + 0x7FFFu + ((u >> 16) & 1u)) >> 16;   // RNE
    return r & 0xFFFFu;
}

__device__ __forceinline__ float sigm(float x) {
    return 1.0f / (1.0f + __expf(-x));
}

// A fragment: 16x32 bf16 tile (rows m0.., cols k0..).
// lane L -> row M=L%16, half=L/16; VGPR g holds K pair 2g+16*half.
__device__ __forceinline__ v16bf load_tileA_bf16(const uint16_t* __restrict__ A,
                                                 int lda, int m0, int k0, int lane) {
    int row  = m0 + (lane & 15);
    int half = lane >> 4;
    const uint32_t* p = (const uint32_t*)(A + (size_t)row * lda);
    v8u u;
#pragma unroll
    for (int g = 0; g < 8; ++g) {
        int kg = 2 * g + 16 * half;
        u[g] = p[(k0 + kg) >> 1];
    }
    return __builtin_bit_cast(v16bf, u);
}

// B fragment: 32x16 bf16 tile (KxN), B stored [N][K] row-major in global.
// lane L -> col N=L%16; lanes 0-15 hold K=k0..k0+15, lanes 16-31 K=k0+16..k0+31.
__device__ __forceinline__ v16bf load_tileB_bf16(const uint16_t* __restrict__ B,
                                                 int ldb, int n0, int k0, int lane) {
    int col  = n0 + (lane & 15);
    int half = lane >> 4;
    const uint32_t* p = (const uint32_t*)(B + (size_t)col * ldb);
    v8u u;
#pragma unroll
    for (int g = 0; g < 8; ++g) u[g] = p[((k0 + half * 16) >> 1) + g];
    return __builtin_bit_cast(v16bf, u);
}

// B fragment from an LDS-staged tile ([32][LDB] bf16, column-major rows).
__device__ __forceinline__ v16bf load_tileB_lds(const uint16_t* ldsB, int ldb_el,
                                                int ncol0, int k0, int lane) {
    const uint32_t* p = (const uint32_t*)(ldsB + (size_t)(ncol0 + (lane & 15)) * ldb_el);
    int half = lane >> 4;
    v8u u;
#pragma unroll
    for (int g = 0; g < 8; ++g) u[g] = p[((k0 + half * 16) >> 1) + g];
    return __builtin_bit_cast(v16bf, u);
}

// ---------- conversion / setup kernels ----------
__global__ void k_f32_to_bf16(const float* __restrict__ in, uint16_t* __restrict__ out, int n) {
    int i = blockIdx.x * blockDim.x + threadIdx.x;
    if (i < n) out[i] = (uint16_t)f32_to_bf16_bits(in[i]);
}

__global__ void k_embed_bf16(const int* __restrict__ tok, const float* __restrict__ embed,
                             uint16_t* __restrict__ out, int L) {
    int i = blockIdx.x * blockDim.x + threadIdx.x;
    if (i >= L * E) return;
    int l = i / E, e = i - l * E;
    out[i] = (uint16_t)f32_to_bf16_bits(embed[(size_t)tok[l] * E + e]);
}

__global__ void k_zero(float* __restrict__ p, int n) {
    int i = blockIdx.x * blockDim.x + threadIdx.x;
    if (i < n) p[i] = 0.0f;
}

// Decoder init: h0 = enc_h, c0 = tanh(enc_h)
__global__ void k_dec_init(const float* __restrict__ enc_h,
                           float* __restrict__ h, float* __restrict__ c) {
    int i = blockIdx.x * blockDim.x + threadIdx.x;
    if (i < H) { float v = enc_h[i]; h[i] = v; c[i] = tanhf(v); }
}

// ---------- WMMA GEMM: Xg[l,j] = emb[l,:]·Wih[j,:] + bih[j] + bhh[j] ----------
__global__ __launch_bounds__(256)
void k_gemm_xg(const uint16_t* __restrict__ A, const uint16_t* __restrict__ B,
               const float* __restrict__ bih, const float* __restrict__ bhh,
               float* __restrict__ C, int M, int N, int K) {
    int wave = (blockIdx.x * blockDim.x + threadIdx.x) >> 5;
    int lane = threadIdx.x & 31;
    int ntn = N >> 4;
    if (wave >= (M >> 4) * ntn) return;
    int m0 = (wave / ntn) << 4;
    int n0 = (wave % ntn) << 4;

    v8f acc = {};
    for (int k0 = 0; k0 < K; k0 += 32) {
        v16bf a = load_tileA_bf16(A, K, m0, k0, lane);
        v16bf b = load_tileB_bf16(B, K, n0, k0, lane);
        acc = __builtin_amdgcn_wmma_f32_16x16x32_bf16(false, a, false, b,
                                                      (short)0, acc, false, false);
    }
    int half = lane >> 4;
    int nc   = n0 + (lane & 15);
    float bias = bih[nc] + bhh[nc];
#pragma unroll
    for (int g = 0; g < 8; ++g) {
        C[(size_t)(m0 + g + 8 * half) * N + nc] = acc[g] + bias;
    }
}

// ---------- persistent LSTM scan ----------
// ONE launch runs all `steps` timesteps. grid = 32 blocks (co-resident),
// cross-block sync via agent-scope atomic counter (release arrive / acquire spin).
// Block b owns n in [32b, 32b+32); computes the 4 gate values for those n.
__global__ __launch_bounds__(256)
void k_lstm_scan(const float* __restrict__ xg, const float* __restrict__ Whh,
                 float* __restrict__ hbuf /* [2][H] */, float* __restrict__ c,
                 uint16_t* __restrict__ hs /* [steps][H] or null */,
                 int steps, int* __restrict__ bar) {
    __shared__ float h_sh[H];
    __shared__ float g_sh[128];

    int tid   = threadIdx.x;
    int wave  = tid >> 5;
    int lane  = tid & 31;
    int nbase = blockIdx.x << 5;

    for (int t = 0; t < steps; ++t) {
        const float* h_prev = hbuf + (size_t)(t & 1) * H;
        float*       h_new  = hbuf + (size_t)((t + 1) & 1) * H;
        const float* xg_t   = xg + (size_t)t * G4H;

        for (int i = tid; i < H; i += 256) h_sh[i] = h_prev[i];
        __syncthreads();

        // 8 waves x 16 dots = 128 = 4 gates x 32 n
        for (int i = 0; i < 16; ++i) {
            int d    = wave * 16 + i;
            int gate = d >> 5;
            int nl   = d & 31;
            int j    = gate * H + nbase + nl;
            const float* row = Whh + (size_t)j * H;
            if (i + 1 < 16) {
                int dn = wave * 16 + i + 1;
                int jn = (dn >> 5) * H + nbase + (dn & 31);
                __builtin_prefetch(Whh + (size_t)jn * H + lane * 4, 0, 1);
            }
            float sum = 0.0f;
#pragma unroll 8
            for (int k = lane; k < H; k += 32) sum += row[k] * h_sh[k];
#pragma unroll
            for (int off = 16; off > 0; off >>= 1) sum += __shfl_xor(sum, off, 32);
            if (lane == 0) g_sh[d] = xg_t[j] + sum;
        }
        __syncthreads();

        if (tid < 32) {
            int   n  = nbase + tid;
            float iv = sigm(g_sh[tid]);
            float fv = sigm(g_sh[32 + tid]);
            float gv = tanhf(g_sh[64 + tid]);
            float ov = sigm(g_sh[96 + tid]);
            float cn = fv * c[n] + iv * gv;
            c[n] = cn;
            float hn = ov * tanhf(cn);
            h_new[n] = hn;
            if (hs) hs[(size_t)t * H + n] = (uint16_t)f32_to_bf16_bits(hn);
        }
        __syncthreads();

        // ---- grid-wide barrier (32 blocks, monotone counter) ----
        if (tid == 0) {
            __threadfence();
            __hip_atomic_fetch_add(bar, 1, __ATOMIC_RELEASE, __HIP_MEMORY_SCOPE_AGENT);
            int want = 32 * (t + 1);
            while (__hip_atomic_load(bar, __ATOMIC_ACQUIRE, __HIP_MEMORY_SCOPE_AGENT) < want)
                __builtin_amdgcn_s_sleep(1);
        }
        __syncthreads();
    }
}

// ---------- fused output projection + softmax partials -----------------------
// Block = 8 waves; block stages ONE 32-column Wout tile (full K) into LDS via
// gfx1250 async global->LDS copies, then every wave computes a 32x32 logit
// region (rows = rowGroup*256 + wave*32) with 4 x v_wmma per k-step, B read
// from LDS (8x reuse). Emits per-row partial max / sum-exp into pm/ps [NVB2][T]
// and captures target logits into z[t].
template <bool PRECONV>
__global__ __launch_bounds__(256)
void k_ce_partial(const uint16_t* __restrict__ hs,
                  const uint16_t* __restrict__ WoutB16,
                  const float* __restrict__ WoutF32,
                  const float* __restrict__ bout, const int* __restrict__ target,
                  float* __restrict__ pm, float* __restrict__ ps, float* __restrict__ z) {
    constexpr int LDB = H + 8;              // padded column stride (elements)
    __shared__ uint16_t ldsB[32 * LDB];     // ~66 KB

    int tid = threadIdx.x;
    int bid = blockIdx.x;
    int rg  = bid / NVB2;                   // row group (256 rows each)
    int vb  = bid % NVB2;
    int v0  = vb << 5;

    // ---- stage B tile: 32 vocab columns x K(=H) bf16 into LDS ----
    {
        int col  = tid >> 3;                // 0..31
        int part = tid & 7;                 // 0..7 : 256B chunk of a 2KB row
        uint16_t* lrow = ldsB + (size_t)col * LDB + part * 128;
        if constexpr (PRECONV) {
            const uint16_t* grow = WoutB16 + (size_t)(v0 + col) * H + part * 128;
            uint32_t laddr = (uint32_t)(uintptr_t)lrow;   // LDS byte offset
#pragma unroll
            for (int ii = 0; ii < 16; ++ii) {
                // ASYNCcnt-tracked DMA: global -> LDS, 16B per lane
                asm volatile("global_load_async_to_lds_b128 %0, %1, off"
                             :: "v"(laddr + ii * 16), "v"(grow + ii * 8)
                             : "memory");
            }
            asm volatile("s_wait_asynccnt 0" ::: "memory");
        } else {
            const float* grow = WoutF32 + (size_t)(v0 + col) * H + part * 128;
#pragma unroll 8
            for (int ii = 0; ii < 128; ++ii)
                lrow[ii] = (uint16_t)f32_to_bf16_bits(grow[ii]);
        }
    }
    __syncthreads();

    int wave = tid >> 5;
    int lane = tid & 31;
    int t0   = rg * 256 + wave * 32;

    v8f acc00 = {}, acc01 = {}, acc10 = {}, acc11 = {};
    for (int k0 = 0; k0 < H; k0 += 32) {
        v16bf a0 = load_tileA_bf16(hs, H, t0,      k0, lane);
        v16bf a1 = load_tileA_bf16(hs, H, t0 + 16, k0, lane);
        v16bf b0 = load_tileB_lds(ldsB, LDB, 0,  k0, lane);
        v16bf b1 = load_tileB_lds(ldsB, LDB, 16, k0, lane);
        acc00 = __builtin_amdgcn_wmma_f32_16x16x32_bf16(false, a0, false, b0, (short)0, acc00, false, false);
        acc01 = __builtin_amdgcn_wmma_f32_16x16x32_bf16(false, a0, false, b1, (short)0, acc01, false, false);
        acc10 = __builtin_amdgcn_wmma_f32_16x16x32_bf16(false, a1, false, b0, (short)0, acc10, false, false);
        acc11 = __builtin_amdgcn_wmma_f32_16x16x32_bf16(false, a1, false, b1, (short)0, acc11, false, false);
    }

    int   half = lane >> 4;
    int   c0   = v0 + (lane & 15);
    int   c1   = c0 + 16;
    float bv0  = bout[c0];
    float bv1  = bout[c1];

#pragma unroll
    for (int i = 0; i < 2; ++i) {
        const v8f& aj = i ? acc10 : acc00;
        const v8f& ak = i ? acc11 : acc01;
        float mx[8], ex[8];
#pragma unroll
        for (int g = 0; g < 8; ++g) {
            float va  = aj[g] + bv0;
            float vbx = ak[g] + bv1;
            int   t   = t0 + 16 * i + g + 8 * half;
            int   tg  = target[t];
            if (tg == c0) z[t] = va;              // unique writer
            if (tg == c1) z[t] = vbx;             // unique writer
            float m = fmaxf(va, vbx);
#pragma unroll
            for (int o = 1; o < 16; o <<= 1) m = fmaxf(m, __shfl_xor(m, o, 32));
            float e = __expf(va - m) + __expf(vbx - m);
#pragma unroll
            for (int o = 1; o < 16; o <<= 1) e += __shfl_xor(e, o, 32);
            mx[g] = m; ex[g] = e;
        }
        if ((lane & 15) == 0) {
#pragma unroll
            for (int g = 0; g < 8; ++g) {
                int t = t0 + 16 * i + g + 8 * half;
                pm[(size_t)vb * T + t] = mx[g];
                ps[(size_t)vb * T + t] = ex[g];
            }
        }
    }
}

// Combine partials: loss[t] = M + log(S) - z[t]
__global__ __launch_bounds__(256)
void k_ce_final(const float* __restrict__ pm, const float* __restrict__ ps,
                const float* __restrict__ z, float* __restrict__ out) {
    int t = blockIdx.x * blockDim.x + threadIdx.x;
    if (t >= T) return;
    float M = -3.4e38f;
    for (int j = 0; j < NVB2; ++j) M = fmaxf(M, pm[(size_t)j * T + t]);
    float Ssum = 0.0f;
    for (int j = 0; j < NVB2; ++j)
        Ssum += ps[(size_t)j * T + t] * __expf(pm[(size_t)j * T + t] - M);
    out[t] = M + logf(Ssum) - z[t];
}

// ---------- host launch ----------
extern "C" void kernel_launch(void* const* d_in, const int* in_sizes, int n_in,
                              void* d_out, int out_size, void* d_ws, size_t ws_size,
                              hipStream_t stream) {
    const int*   source   = (const int*)  d_in[0];
    const int*   target   = (const int*)  d_in[1];
    const float* encEmbed = (const float*)d_in[2];
    const float* encWih   = (const float*)d_in[3];
    const float* encWhh   = (const float*)d_in[4];
    const float* encBih   = (const float*)d_in[5];
    const float* encBhh   = (const float*)d_in[6];
    const float* decEmbed = (const float*)d_in[7];
    const float* decWih   = (const float*)d_in[8];
    const float* decWhh   = (const float*)d_in[9];
    const float* decBih   = (const float*)d_in[10];
    const float* decBhh   = (const float*)d_in[11];
    const float* Wout     = (const float*)d_in[12];
    const float* bout     = (const float*)d_in[13];
    float* out = (float*)d_out;

    // workspace carve-out (256B aligned slabs)
    char*  base = (char*)d_ws;
    size_t off  = 0;
    auto alloc = [&](size_t bytes) -> void* {
        off = (off + 255) & ~(size_t)255;
        void* p = base + off;
        off += bytes;
        return p;
    };
    float*    xgE   = (float*)   alloc((size_t)S * G4H * 4);    // 8 MB
    float*    xgD   = (float*)   alloc((size_t)T * G4H * 4);    // 8 MB
    float*    pm    = (float*)   alloc((size_t)NVB2 * T * 4);   // 2 MB
    float*    ps    = (float*)   alloc((size_t)NVB2 * T * 4);   // 2 MB
    float*    zbuf  = (float*)   alloc((size_t)T * 4);
    float*    hE    = (float*)   alloc((size_t)2 * H * 4);      // double buffer (8 KB)
    float*    cE    = (float*)   alloc((size_t)H * 4);          // 4 KB
    int*      barE  = (int*)     alloc(256);
    int*      barD  = (int*)     alloc(256);
    float*    hD    = (float*)   alloc((size_t)2 * H * 4);
    float*    cD    = (float*)   alloc((size_t)H * 4);
    uint16_t* embS  = (uint16_t*)alloc((size_t)S * E * 2);
    uint16_t* embT  = (uint16_t*)alloc((size_t)T * E * 2);
    uint16_t* WihEb = (uint16_t*)alloc((size_t)G4H * E * 2);    // 4 MB
    uint16_t* WihDb = (uint16_t*)alloc((size_t)G4H * E * 2);    // 4 MB
    uint16_t* hsb   = (uint16_t*)alloc((size_t)T * H * 2);      // 1 MB

    // Optional bf16 copy of Wout (64 MB) if workspace allows.
    uint16_t* Woutb = nullptr;
    {
        size_t need = ((off + 255) & ~(size_t)255) + (size_t)V * H * 2;
        if (need <= ws_size) Woutb = (uint16_t*)alloc((size_t)V * H * 2);
    }
    (void)n_in; (void)in_sizes; (void)out_size;

    // 1) embeddings -> bf16 ; Wih -> bf16 ; zero encoder state + barriers
    k_embed_bf16<<<(S * E + 255) / 256, 256, 0, stream>>>(source, encEmbed, embS, S);
    k_embed_bf16<<<(T * E + 255) / 256, 256, 0, stream>>>(target, decEmbed, embT, T);
    k_f32_to_bf16<<<(G4H * E + 255) / 256, 256, 0, stream>>>(encWih, WihEb, G4H * E);
    k_f32_to_bf16<<<(G4H * E + 255) / 256, 256, 0, stream>>>(decWih, WihDb, G4H * E);
    // hE[2H], cE[H], barE(64 fl), barD(64 fl) are contiguous 256B slabs -> one zero pass
    k_zero<<<(3 * H + 128 + 255) / 256, 256, 0, stream>>>(hE, 3 * H + 128);
    if (Woutb)
        k_f32_to_bf16<<<(V * H + 255) / 256, 256, 0, stream>>>(Wout, Woutb, V * H);

    // 2) batched input-gate GEMMs (WMMA): xg = emb @ Wih.T + bih + bhh
    {
        int waves = (S / 16) * (G4H / 16);            // 8192
        k_gemm_xg<<<(waves * 32 + 255) / 256, 256, 0, stream>>>(
            embS, WihEb, encBih, encBhh, xgE, S, G4H, E);
        k_gemm_xg<<<(waves * 32 + 255) / 256, 256, 0, stream>>>(
            embT, WihDb, decBih, decBhh, xgD, T, G4H, E);
    }

    // 3) persistent encoder scan (single launch, grid barrier inside)
    k_lstm_scan<<<H / 32, 256, 0, stream>>>(xgE, encWhh, hE, cE, nullptr, S, barE);
    // S even -> final encoder h lives in hE[0]
    k_dec_init<<<(H + 255) / 256, 256, 0, stream>>>(hE + (size_t)(S & 1) * H, hD, cD);

    // 4) persistent decoder scan, emitting hs rows as bf16
    k_lstm_scan<<<H / 32, 256, 0, stream>>>(xgD, decWhh, hD, cD, hsb, T, barD);

    // 5) fused output projection (WMMA 2x2, LDS-staged B) + softmax partials + CE
    {
        int blocks = (T / 256) * NVB2;                // 2000
        if (Woutb)
            k_ce_partial<true><<<blocks, 256, 0, stream>>>(
                hsb, Woutb, Wout, bout, target, pm, ps, zbuf);
        else
            k_ce_partial<false><<<blocks, 256, 0, stream>>>(
                hsb, Woutb, Wout, bout, target, pm, ps, zbuf);
        k_ce_final<<<(T + 255) / 256, 256, 0, stream>>>(pm, ps, zbuf, out);
    }
}
